// TransformerEncoder_1838246002911
// MI455X (gfx1250) — matrix operand order repaired and checked by reading, not compiled, not run on hardware
//
#include <hip/hip_runtime.h>
#include <hip/hip_bf16.h>

typedef __bf16 bf16;
typedef __attribute__((ext_vector_type(16))) __bf16 v16bf;
typedef __attribute__((ext_vector_type(8)))  float  v8f;
typedef int v4i __attribute__((vector_size(16)));

#if defined(__has_builtin)
#if __has_builtin(__builtin_amdgcn_global_load_async_to_lds_b128)
#define HAVE_ASYNC_LDS 1
#endif
#endif
#ifndef HAVE_ASYNC_LDS
#define HAVE_ASYNC_LDS 0
#endif

// 16-byte global -> LDS copy via global_load_async_to_lds_b128 (ASYNCcnt).
static __device__ __forceinline__ void g2l16(const bf16* g, bf16* l) {
#if HAVE_ASYNC_LDS
  __builtin_amdgcn_global_load_async_to_lds_b128(
      (__attribute__((address_space(1))) v4i*)g,
      (__attribute__((address_space(3))) v4i*)l, 0, 0);
#else
  *(uint4*)l = *(const uint4*)g;
#endif
}
static __device__ __forceinline__ void g2l_wait() {
#if HAVE_ASYNC_LDS
  asm volatile("s_wait_asynccnt 0" ::: "memory");
#endif
}

static __device__ __forceinline__ v8f wmma_bf16f32(v16bf a, v16bf b, v8f c) {
  // D = A(16x32 bf16) * B(32x16 bf16) + C(16x16 f32)
  return __builtin_amdgcn_wmma_f32_16x16x32_bf16(false, a, false, b, (short)0, c,
                                                 false, false);
}

static __device__ __forceinline__ float hmax16(float v) {
  v = fmaxf(v, __shfl_xor(v, 1, 32));
  v = fmaxf(v, __shfl_xor(v, 2, 32));
  v = fmaxf(v, __shfl_xor(v, 4, 32));
  v = fmaxf(v, __shfl_xor(v, 8, 32));
  return v;
}
static __device__ __forceinline__ float hsum16(float v) {
  v += __shfl_xor(v, 1, 32);
  v += __shfl_xor(v, 2, 32);
  v += __shfl_xor(v, 4, 32);
  v += __shfl_xor(v, 8, 32);
  return v;
}

// ---------------------------------------------------------------------------
// fp32 -> bf16 staging (x) and fp32 -> bf16 transpose staging (weights).
// cvtT: dst[N][K] = src[K][N] * sc   (32x32 LDS-tiled)
// ---------------------------------------------------------------------------
__global__ void cvt_f32_to_bf16(const float* __restrict__ src,
                                bf16* __restrict__ dst, int n, float sc) {
  int i = blockIdx.x * blockDim.x + threadIdx.x;
  const int stride = gridDim.x * blockDim.x;
  for (; i < n; i += stride) dst[i] = (bf16)(src[i] * sc);
}
__global__ __launch_bounds__(256) void cvtT_f32_to_bf16(
    const float* __restrict__ src, bf16* __restrict__ dst, int K, int N,
    float sc) {
  __shared__ float tl[32][33];
  const int n0 = blockIdx.x * 32, k0 = blockIdx.y * 32;
  for (int i = threadIdx.y; i < 32; i += 8)
    tl[i][threadIdx.x] = src[(size_t)(k0 + i) * N + n0 + threadIdx.x];
  __syncthreads();
  for (int i = threadIdx.y; i < 32; i += 8)
    dst[(size_t)(n0 + i) * K + k0 + threadIdx.x] =
        (bf16)(tl[threadIdx.x][i] * sc);
}
__global__ void scale_copy_f32(const float* __restrict__ src,
                               float* __restrict__ dst, int n, float sc) {
  const int i = blockIdx.x * blockDim.x + threadIdx.x;
  if (i < n) dst[i] = src[i] * sc;
}

// ---------------------------------------------------------------------------
// Tiled bf16 WMMA GEMM, async-LDS double buffered, weights pre-transposed:
// C[M,N] = A[M,K] @ B[K,N] + bias (+ ReLU), with BT = B^T stored [N,K].
// Both LDS tiles are [128][32] so every fragment read is contiguous b128s.
// OUT_MODE: 0 f32 [M,N], 1 bf16 [M,N], 2 bf16 [B,H,S,HD], 3 bf16 [B,H,HD,S]
// ---------------------------------------------------------------------------
template <int OUT_MODE, bool RELU>
__global__ __launch_bounds__(256) void gemm_bf16_wmma(
    const bf16* __restrict__ A, const bf16* __restrict__ BT,
    const float* __restrict__ bias, void* __restrict__ Cout,
    int M, int N, int K, int Sdim, int HDdim) {
  __shared__ __align__(16) bf16 As[2][128][32];
  __shared__ __align__(16) bf16 BsT[2][128][32];

  const int t = threadIdx.x;
  const int wave = t >> 5, lane = t & 31;
  const int lh = (lane >> 4) & 1, l15 = lane & 15;
  const int wm = (wave >> 2) * 64;
  const int wn = (wave & 3) * 32;
  const int m0 = blockIdx.y * 128;
  const int n0 = blockIdx.x * 128;

  auto stage = [&](int buf, int k0) {
#pragma unroll
    for (int i = 0; i < 2; ++i) {  // 256 thr * 2 * 16B per matrix
      const int c = t + (i << 8);
      const int r = c >> 2, cc = (c & 3) << 3;
      g2l16(A + (size_t)(m0 + r) * K + k0 + cc, &As[buf][r][cc]);
      g2l16(BT + (size_t)(n0 + r) * K + k0 + cc, &BsT[buf][r][cc]);
    }
  };

  v8f acc[4][2] = {};
  const int nk = K >> 5;
  stage(0, 0);

  for (int ki = 0; ki < nk; ++ki) {
    const int buf = ki & 1;
    g2l_wait();        // our async tile writes done
    __syncthreads();   // everyone's writes done; prev reads of buf^1 done
    if (ki + 1 < nk) stage(buf ^ 1, (ki + 1) << 5);

    // A fragment: lanes 0-15 K 0..7,16..23; lanes 16-31 +8 (contiguous pairs)
    v16bf af[4];
#pragma unroll
    for (int i = 0; i < 4; ++i) {
      const int m = wm + i * 16 + l15;
      union { v16bf v; unsigned u[8]; } fr;
#pragma unroll
      for (int p = 0; p < 8; ++p) {
        const int kk = ((p < 4) ? (2 * p) : (16 + 2 * (p - 4))) + (lh ? 8 : 0);
        fr.u[p] = *(const unsigned*)&As[buf][m][kk];
      }
      af[i] = fr.v;
    }
    // B fragment: lane = n; low lanes K 0..7,16..23; high lanes +8; element
    // B[k][n] = BT[n][k] -> contiguous pairs in BsT row n.
    v16bf bfr[2];
#pragma unroll
    for (int j = 0; j < 2; ++j) {
      const int n = wn + j * 16 + l15;
      const int koff = lh ? 8 : 0;
      union { v16bf v; unsigned u[8]; } fr;
#pragma unroll
      for (int p = 0; p < 8; ++p)
        fr.u[p] = *(const unsigned*)&BsT[buf][n][koff + ((p < 4) ? 2 * p : 16 + 2 * (p - 4))];
      bfr[j] = fr.v;
    }
#pragma unroll
    for (int i = 0; i < 4; ++i)
#pragma unroll
      for (int j = 0; j < 2; ++j)
        acc[i][j] = wmma_bf16f32(af[i], bfr[j], acc[i][j]);
  }

  const int Hn = (OUT_MODE >= 2) ? (N / HDdim) : 0;
#pragma unroll
  for (int i = 0; i < 4; ++i) {
#pragma unroll
    for (int j = 0; j < 2; ++j) {
      const int n = n0 + wn + j * 16 + l15;
      const float bv = bias[n];
#pragma unroll
      for (int r = 0; r < 8; ++r) {
        const int m = m0 + wm + i * 16 + r + (lh ? 8 : 0);
        float v = acc[i][j][r] + bv;
        if (RELU) v = fmaxf(v, 0.0f);
        if (OUT_MODE == 0) {
          ((float*)Cout)[(size_t)m * N + n] = v;
        } else if (OUT_MODE == 1) {
          ((bf16*)Cout)[(size_t)m * N + n] = (bf16)v;
        } else {
          const int b = m / Sdim, s = m - b * Sdim;
          const int h = n / HDdim, hd = n - h * HDdim;
          const size_t idx =
              (OUT_MODE == 2)
                  ? ((((size_t)b * Hn + h) * Sdim + s) * HDdim + hd)
                  : ((((size_t)b * Hn + h) * HDdim + hd) * Sdim + s);
          ((bf16*)Cout)[idx] = (bf16)v;
        }
      }
    }
  }
}

// ---------------------------------------------------------------------------
// Flash attention: one (b,h) per blockIdx.x, 64 query rows per blockIdx.y,
// 4 waves x 16 query rows. Q pre-scaled by 1/sqrt(HD) (folded into qW).
// K is [B,H,S,HD]; V is pre-transposed [B,H,HD,S] so the P.V B-fragment is
// contiguous. K/V stream async-LDS double buffered; both matmuls via WMMA.
// ---------------------------------------------------------------------------
__global__ __launch_bounds__(128) void flash_attn_wmma(
    const bf16* __restrict__ Q, const bf16* __restrict__ Kh,
    const bf16* __restrict__ VhT, const int* __restrict__ lengths,
    bf16* __restrict__ ctx, int Bn, int Hn, int Sn, int HDn) {
  const int bh = blockIdx.x;
  const int b = bh / Hn, h = bh - b * Hn;
  const int q0 = blockIdx.y * 64;
  const int t = threadIdx.x, wave = t >> 5, lane = t & 31;
  const int lh = (lane >> 4) & 1, l15 = lane & 15;
  const int len = lengths[b];

  __shared__ __align__(16) bf16 Qs[64][64];
  __shared__ __align__(16) bf16 Ks[2][64][64];    // [key][hd]
  __shared__ __align__(16) bf16 VsT[2][64][64];   // [hd][key]
  __shared__ __align__(16) bf16 Ps[4][16][64];

  const size_t hoff = (size_t)bh * Sn * HDn;

  auto stageKV = [&](int buf, int k0) {
#pragma unroll
    for (int i = 0; i < 4; ++i) {  // 128 thr * 4 * 16B per 64x64 tile
      const int c = t + (i << 7);
      const int r = c >> 3, cc = (c & 7) << 3;
      g2l16(Kh + hoff + (size_t)(k0 + r) * HDn + cc, &Ks[buf][r][cc]);
      g2l16(VhT + hoff + (size_t)r * Sn + k0 + cc, &VsT[buf][r][cc]);
    }
  };

  // stage Q and first K/V block together
#pragma unroll
  for (int i = 0; i < 4; ++i) {
    const int c = t + (i << 7);
    const int r = c >> 3, cc = (c & 7) << 3;
    g2l16(Q + hoff + (size_t)(q0 + r) * HDn + cc, &Qs[r][cc]);
  }
  const int nkb = (len + 63) >> 6;
  stageKV(0, 0);
  g2l_wait();
  __syncthreads();

  // Q A-fragments (two 32-wide contraction chunks of HD=64)
  v16bf qf[2];
#pragma unroll
  for (int c = 0; c < 2; ++c) {
    union { v16bf v; unsigned u[8]; } fr;
    const int m = wave * 16 + l15;
#pragma unroll
    for (int p = 0; p < 8; ++p) {
      const int kk =
          c * 32 + ((p < 4) ? 2 * p : 16 + 2 * (p - 4)) + (lh ? 8 : 0);
      fr.u[p] = *(const unsigned*)&Qs[m][kk];
    }
    qf[c] = fr.v;
  }

  v8f acc[4] = {};
  float mrow[8], lrow[8];
#pragma unroll
  for (int r = 0; r < 8; ++r) { mrow[r] = -3.0e38f; lrow[r] = 0.0f; }

  for (int kb = 0; kb < nkb; ++kb) {
    const int k0 = kb * 64;
    const int buf = kb & 1;
    if (kb > 0) {
      g2l_wait();
      __syncthreads();
    }
    if (kb + 1 < nkb) stageKV(buf ^ 1, k0 + 64);

    // logits = Q K^T : B element (k=hd, n=key) = Ks[key][hd] (contiguous)
    v8f sc[4] = {};
#pragma unroll
    for (int jt = 0; jt < 4; ++jt) {
      const int n = jt * 16 + l15;
#pragma unroll
      for (int c = 0; c < 2; ++c) {
        union { v16bf v; unsigned u[8]; } kf;
        const int koff = c * 32 + (lh ? 8 : 0);
#pragma unroll
        for (int p = 0; p < 8; ++p)
          kf.u[p] = *(const unsigned*)&Ks[buf][n][koff + ((p < 4) ? 2 * p : 16 + 2 * (p - 4))];
        sc[jt] = wmma_bf16f32(qf[c], kf.v, sc[jt]);
      }
    }

    // key mask + online softmax (Q already carries the 1/sqrt(HD) scale)
    const bool maskblk = (k0 + 64 > len);
    float bmax[8];
#pragma unroll
    for (int r = 0; r < 8; ++r) bmax[r] = -3.0e38f;
#pragma unroll
    for (int jt = 0; jt < 4; ++jt) {
      const int n = k0 + jt * 16 + l15;
      const bool oob = maskblk && (n >= len);
#pragma unroll
      for (int r = 0; r < 8; ++r) {
        float v = sc[jt][r];
        if (oob) v = -3.0e38f;
        sc[jt][r] = v;
        bmax[r] = fmaxf(bmax[r], v);
      }
    }
#pragma unroll
    for (int r = 0; r < 8; ++r) {
      const float mnew = fmaxf(mrow[r], hmax16(bmax[r]));
      const float corr = __expf(mrow[r] - mnew);
      mrow[r] = mnew;
      lrow[r] *= corr;
#pragma unroll
      for (int j = 0; j < 4; ++j) acc[j][r] *= corr;
    }
#pragma unroll
    for (int r = 0; r < 8; ++r) {
      float ps = 0.0f;
#pragma unroll
      for (int jt = 0; jt < 4; ++jt) {
        const float p = __expf(sc[jt][r] - mrow[r]);
        sc[jt][r] = p;
        ps += p;
      }
      lrow[r] += hsum16(ps);
    }

    // C-layout -> A-layout bounce of P through per-wave LDS
#pragma unroll
    for (int jt = 0; jt < 4; ++jt)
#pragma unroll
      for (int r = 0; r < 8; ++r)
        Ps[wave][r + (lh ? 8 : 0)][jt * 16 + l15] = (bf16)sc[jt][r];
    asm volatile("s_wait_dscnt 0" ::: "memory");  // wave-local LDS fence

    // O += P.V : B element (k=key, n=hd) = VsT[hd][key] (contiguous pairs)
#pragma unroll
    for (int c = 0; c < 2; ++c) {
      union { v16bf v; unsigned u[8]; } pf;
#pragma unroll
      for (int p = 0; p < 8; ++p) {
        const int kk =
            c * 32 + ((p < 4) ? 2 * p : 16 + 2 * (p - 4)) + (lh ? 8 : 0);
        pf.u[p] = *(const unsigned*)&Ps[wave][l15][kk];
      }
#pragma unroll
      for (int j = 0; j < 4; ++j) {
        const int n = j * 16 + l15;
        const int koff = c * 32 + (lh ? 8 : 0);
        union { v16bf v; unsigned u[8]; } vf;
#pragma unroll
        for (int p = 0; p < 8; ++p)
          vf.u[p] = *(const unsigned*)&VsT[buf][n][koff + ((p < 4) ? 2 * p : 16 + 2 * (p - 4))];
        acc[j] = wmma_bf16f32(pf.v, vf.v, acc[j]);
      }
    }
  }

  // write ctx in [B,S,D] layout, bf16 for the O-projection GEMM
  const int Dn = Hn * HDn;
#pragma unroll
  for (int j = 0; j < 4; ++j) {
    const int hd = j * 16 + l15;
#pragma unroll
    for (int r = 0; r < 8; ++r) {
      const int s = q0 + wave * 16 + r + (lh ? 8 : 0);
      const float ov = acc[j][r] / lrow[r];
      ctx[(size_t)(b * Sn + s) * Dn + h * HDn + hd] = (bf16)ov;
    }
  }
}

// ---------------------------------------------------------------------------
extern "C" void kernel_launch(void* const* d_in, const int* in_sizes, int n_in,
                              void* d_out, int out_size, void* d_ws,
                              size_t ws_size, hipStream_t stream) {
  (void)in_sizes; (void)n_in; (void)out_size; (void)ws_size;
  constexpr int Bn = 4, Sn = 2048, Dn = 1024, Hn = 16, FFn = 4096, HDn = 64;
  const int M = Bn * Sn;  // 8192
  const float att_scale = 0.125f;  // 1/sqrt(HD)

  const float* x     = (const float*)d_in[0];
  const int*   lens  = (const int*)d_in[1];
  const float* qW    = (const float*)d_in[2];
  const float* qb    = (const float*)d_in[3];
  const float* kW    = (const float*)d_in[4];
  const float* kbias = (const float*)d_in[5];
  const float* vW    = (const float*)d_in[6];
  const float* vb    = (const float*)d_in[7];
  const float* oW    = (const float*)d_in[8];
  const float* ob    = (const float*)d_in[9];
  const float* w1    = (const float*)d_in[10];
  const float* b1    = (const float*)d_in[11];
  const float* w2    = (const float*)d_in[12];
  const float* b2    = (const float*)d_in[13];

  size_t off = 0;
  auto bump = [&](size_t bytes) -> void* {
    void* p = (char*)d_ws + off;
    off += (bytes + 255) & ~(size_t)255;
    return p;
  };
  bf16*  xb   = (bf16*)bump((size_t)M * Dn * 2);
  bf16*  qWt  = (bf16*)bump((size_t)Dn * Dn * 2);   // all weights stored W^T
  bf16*  kWt  = (bf16*)bump((size_t)Dn * Dn * 2);
  bf16*  vWt  = (bf16*)bump((size_t)Dn * Dn * 2);
  bf16*  oWt  = (bf16*)bump((size_t)Dn * Dn * 2);
  bf16*  w1t  = (bf16*)bump((size_t)Dn * FFn * 2);
  bf16*  w2t  = (bf16*)bump((size_t)FFn * Dn * 2);
  float* qbs  = (float*)bump((size_t)Dn * 4);       // scaled q bias (f32)
  bf16*  qh   = (bf16*)bump((size_t)M * Dn * 2);    // [B,H,S,HD]
  bf16*  kh   = (bf16*)bump((size_t)M * Dn * 2);    // [B,H,S,HD]
  bf16*  vht  = (bf16*)bump((size_t)M * Dn * 2);    // [B,H,HD,S]
  bf16*  ctxb = (bf16*)bump((size_t)M * Dn * 2);    // [B,S,D]
  bf16*  aob  = (bf16*)bump((size_t)M * Dn * 2);    // attn_out bf16
  bf16*  hb   = qh;  // FFN hidden (64 MB) aliases dead qh/kh/vht/ctxb region

  const dim3 cb(256);
  const dim3 tb(32, 8);
  cvt_f32_to_bf16<<<dim3(2048), cb, 0, stream>>>(x, xb, M * Dn, 1.0f);
  cvtT_f32_to_bf16<<<dim3(Dn / 32, Dn / 32), tb, 0, stream>>>(qW, qWt, Dn, Dn,
                                                              att_scale);
  cvtT_f32_to_bf16<<<dim3(Dn / 32, Dn / 32), tb, 0, stream>>>(kW, kWt, Dn, Dn,
                                                              1.0f);
  cvtT_f32_to_bf16<<<dim3(Dn / 32, Dn / 32), tb, 0, stream>>>(vW, vWt, Dn, Dn,
                                                              1.0f);
  cvtT_f32_to_bf16<<<dim3(Dn / 32, Dn / 32), tb, 0, stream>>>(oW, oWt, Dn, Dn,
                                                              1.0f);
  cvtT_f32_to_bf16<<<dim3(FFn / 32, Dn / 32), tb, 0, stream>>>(w1, w1t, Dn, FFn,
                                                               1.0f);
  cvtT_f32_to_bf16<<<dim3(Dn / 32, FFn / 32), tb, 0, stream>>>(w2, w2t, FFn, Dn,
                                                               1.0f);
  scale_copy_f32<<<dim3(Dn / 256), cb, 0, stream>>>(qb, qbs, Dn, att_scale);

  const dim3 gblk(256);
  // QKV projections (q pre-scaled); v scattered transposed to [B,H,HD,S]
  gemm_bf16_wmma<2, false><<<dim3(Dn / 128, M / 128), gblk, 0, stream>>>(
      xb, qWt, qbs, qh, M, Dn, Dn, Sn, HDn);
  gemm_bf16_wmma<2, false><<<dim3(Dn / 128, M / 128), gblk, 0, stream>>>(
      xb, kWt, kbias, kh, M, Dn, Dn, Sn, HDn);
  gemm_bf16_wmma<3, false><<<dim3(Dn / 128, M / 128), gblk, 0, stream>>>(
      xb, vWt, vb, vht, M, Dn, Dn, Sn, HDn);

  flash_attn_wmma<<<dim3(Bn * Hn, Sn / 64), dim3(128), 0, stream>>>(
      qh, kh, vht, lens, ctxb, Bn, Hn, Sn, HDn);

  gemm_bf16_wmma<1, false><<<dim3(Dn / 128, M / 128), gblk, 0, stream>>>(
      ctxb, oWt, ob, aob, M, Dn, Dn, 1, 1);
  gemm_bf16_wmma<1, true><<<dim3(FFn / 128, M / 128), gblk, 0, stream>>>(
      aob, w1t, b1, hb, M, FFn, Dn, 1, 1);
  gemm_bf16_wmma<0, false><<<dim3(Dn / 128, M / 128), gblk, 0, stream>>>(
      hb, w2t, b2, d_out, M, Dn, FFn, 1, 1);
}
